// QuasiRNNDecoder_74148315398213
// MI455X (gfx1250) — matrix operand order repaired and checked by reading, not compiled, not run on hardware
//
#include <hip/hip_runtime.h>
#include <math.h>

// ---------------- problem constants ----------------
#define B_SZ   32
#define T_SZ   2000
#define D_SZ   256
#define H_SZ   512
#define G_SZ   1536      // 3*H
#define C_SZ   41
#define TP_SZ  497       // (2000-14)/4 + 1
#define KLEN_  14
#define F0_SZ  3584      // D * KLEN
#define NDAYS  24

typedef __bf16 bf16;
typedef __attribute__((ext_vector_type(16))) __bf16 v16bf;
typedef __attribute__((ext_vector_type(8)))  float  v8f;

#define BM 128
#define BN 64
#define BK 32

// ---------------- TDM (Tensor Data Mover) async tile load --------------------
#if defined(__HIP_DEVICE_COMPILE__) && __has_builtin(__builtin_amdgcn_tensor_load_to_lds)
#define HAVE_TDM 1
#else
#define HAVE_TDM 0
#endif

#if HAVE_TDM
typedef unsigned int u32x4 __attribute__((ext_vector_type(4)));
typedef int          i32x4 __attribute__((ext_vector_type(4)));
typedef int          i32x8 __attribute__((ext_vector_type(8)));

// 2D tile DMA: tile_n rows x tile_k bf16 elems, row stride = tensor_k elems,
// written contiguously to LDS (row-major [n][k]) -> fragment-ready layout.
// D# packing per CDNA5 ISA 8.3/8.4 (group0 128b, group1 256b; groups 2/3 zero
// for <=2D tensors).
__device__ __forceinline__ void tdm_load_tile_2d(unsigned lds_byte_addr,
                                                 const bf16* gsrc,
                                                 unsigned tile_k, unsigned tile_n,
                                                 unsigned tensor_k, unsigned tensor_n) {
  unsigned long long ga = (unsigned long long)(uintptr_t)gsrc;
  u32x4 g0;
  g0[0] = 1u;                                       // count=1 (valid), user mode
  g0[1] = lds_byte_addr;                            // D#.lds_addr  [63:32]
  g0[2] = (unsigned)ga;                             // global_addr[31:0]
  g0[3] = (unsigned)((ga >> 32) & 0x01FFFFFFu) | (2u << 30);  // addr[56:32] | type=2
  i32x8 g1;
  g1[0] = 1 << 16;                                  // wg_mask=0, data_size=1 (2B)
  g1[1] = (int)(tensor_k << 16);                    // tensor_dim0[15:0]  @[63:48]
  g1[2] = (int)((tensor_k >> 16) | (tensor_n << 16));  // td0[31:16] | td1[15:0]
  g1[3] = (int)((tensor_n >> 16) | (tile_k << 16));    // td1[31:16] | tile_dim0
  g1[4] = (int)(tile_n & 0xFFFF);                   // tile_dim1 (tile_dim2=0)
  g1[5] = (int)tensor_k;                            // tensor_dim0_stride[31:0]
  g1[6] = 0;                                        // stride hi / dim1_stride lo
  g1[7] = 0;
  i32x4 gz = {0, 0, 0, 0};
#if __clang_major__ >= 23
  i32x8 gz8 = {0, 0, 0, 0, 0, 0, 0, 0};
  __builtin_amdgcn_tensor_load_to_lds(g0, g1, gz, gz, gz8, 0);
#else
  __builtin_amdgcn_tensor_load_to_lds(g0, g1, gz, gz, 0);
#endif
}
#endif

// ---------------- WMMA fragment helpers (ISA 7.12.2 layouts, wave32) ----------
__device__ __forceinline__ v16bf frag_a(const bf16* sm, int mbase) {
  int lane = threadIdx.x & 31;
  const bf16* r = sm + (size_t)(mbase + (lane & 15)) * BK + (((lane >> 4) & 1) << 3);
  v16bf a;
#pragma unroll
  for (int e = 0; e < 8; ++e) { a[e] = r[e]; a[e + 8] = r[16 + e]; }
  return a;
}
__device__ __forceinline__ v16bf frag_b(const bf16* sm, int nbase) {
  int lane = threadIdx.x & 31;
  const bf16* r = sm + (size_t)(nbase + (lane & 15)) * BK + (((lane >> 4) & 1) << 3);
  v16bf b;
#pragma unroll
  for (int e = 0; e < 8; ++e) { b[e] = r[e]; b[e + 8] = r[16 + e]; }
  return b;
}

// fast activations (v_rcp_f32 / v_exp_f32)
__device__ __forceinline__ float fast_sigmoid(float x) {
  return __builtin_amdgcn_rcpf(1.f + __expf(-x));
}
__device__ __forceinline__ float fast_tanh(float x) {
  return 2.f * __builtin_amdgcn_rcpf(1.f + __expf(-2.f * x)) - 1.f;
}
__device__ __forceinline__ float fast_softsign(float x) {
  return x * __builtin_amdgcn_rcpf(1.f + __builtin_fabsf(x));
}

__device__ __forceinline__ float wave_red(float v) {
#pragma unroll
  for (int o = 16; o > 0; o >>= 1) v += __shfl_xor(v, o, 32);
  return v;
}

// ---------------- kernel 0a: fp32 -> bf16 convert -----------------------------
__global__ __launch_bounds__(256) void cvt_bf16_kernel(const float* __restrict__ s,
                                                       bf16* __restrict__ d, int n) {
  int i = blockIdx.x * 256 + threadIdx.x;
  if (i < n) d[i] = (bf16)s[i];
}

// ---------------- kernel 0b: dayWeights fp32 -> bf16 + transpose --------------
__global__ __launch_bounds__(256) void cvt_wdT_kernel(const float* __restrict__ s,
                                                      bf16* __restrict__ d) {
  int i = blockIdx.x * 256 + threadIdx.x;          // NDAYS*D*D total
  int day = i >> 16;
  int r = i & 0xFFFF;
  int row = r >> 8;       // d
  int col = r & 255;      // kout
  d[((size_t)day << 16) + ((size_t)col << 8) + row] = (bf16)s[i];
}

// ---------------- kernel 1: gaussian smooth (depthwise, K=20, pad 9/10) -------
__global__ __launch_bounds__(256) void gauss_kernel(const float* __restrict__ in,
                                                    bf16* __restrict__ xs) {
  int bt = blockIdx.x;                // b*T + t
  int d  = threadIdx.x;               // 0..255
  int b = bt / T_SZ, t = bt - b * T_SZ;
  float g[20]; float s = 0.f;
#pragma unroll
  for (int j = 0; j < 20; ++j) {
    float u = ((float)j - 9.5f) * 0.5f;            // (i - 9.5)/SIGMA, SIGMA=2
    g[j] = __expf(-0.5f * u * u); s += g[j];
  }
  float inv = __builtin_amdgcn_rcpf(s);
  float acc = 0.f;
  const float* base = in + (size_t)b * T_SZ * D_SZ + d;
#pragma unroll
  for (int j = 0; j < 20; ++j) {
    int tt = t - 9 + j;
    if (tt >= 0 && tt < T_SZ) acc += base[(size_t)tt * D_SZ] * g[j];
  }
  xs[((size_t)b * T_SZ + t) * D_SZ + d] = (bf16)(acc * inv);
}

// ============== shared GEMM micro-kernel pieces ===============================
// 256 threads = 8 waves; block tile BM x BN; wave w owns a 32x32 micro-tile
struct WaveCtx { int mq, nq, lcol, mrow; };
__device__ __forceinline__ WaveCtx wave_ctx() {
  int wv = threadIdx.x >> 5, lane = threadIdx.x & 31;
  WaveCtx c;
  c.mq = (wv >> 1) * 32;
  c.nq = (wv & 1) * 32;
  c.lcol = lane & 15;
  c.mrow = ((lane >> 4) & 1) * 8;
  return c;
}

// ---------------- kernel 2: per-day GEMM + softsign, transposed store ---------
__global__ __launch_bounds__(256) void day_gemm_kernel(const bf16* __restrict__ xs,
                                                       const bf16* __restrict__ wdT_bf,
                                                       const float* __restrict__ day_bias,
                                                       const int* __restrict__ day_idx,
                                                       bf16* __restrict__ x2t) {
  __shared__ __align__(16) bf16 smA[BM * BK];
  __shared__ __align__(16) bf16 smB[BN * BK];
  const int t0 = blockIdx.x * BM, n0 = blockIdx.y * BN, b = blockIdx.z;
  const int day = day_idx[b];
  const int tid = threadIdx.x;
  WaveCtx wc = wave_ctx();
  v8f a00 = {0.f,0.f,0.f,0.f,0.f,0.f,0.f,0.f};
  v8f a01 = a00, a10 = a00, a11 = a00;
  const bf16* xb = xs + (size_t)b * T_SZ * D_SZ;
  const bf16* Wt = wdT_bf + ((size_t)day << 16);
  const int ar = tid >> 1, ac = (tid & 1) * 16;
  const int brn = tid >> 2, bkg = (tid & 3) * 8;
  for (int k0 = 0; k0 < D_SZ; k0 += BK) {
    int t = t0 + ar;
    uint4 v0 = {0u,0u,0u,0u}, v1 = {0u,0u,0u,0u};
    if (t < T_SZ) {
      const uint4* src = (const uint4*)(xb + (size_t)t * D_SZ + k0 + ac);
      v0 = src[0]; v1 = src[1];
    }
    uint4* dstA = (uint4*)&smA[ar * BK + ac];
    dstA[0] = v0; dstA[1] = v1;
    *(uint4*)&smB[brn * BK + bkg] =
        *(const uint4*)(Wt + (size_t)(n0 + brn) * D_SZ + k0 + bkg);
    __syncthreads();
    v16bf b0 = frag_b(smB, wc.nq);
    v16bf b1 = frag_b(smB, wc.nq + 16);
    v16bf fa0 = frag_a(smA, wc.mq);
    v16bf fa1 = frag_a(smA, wc.mq + 16);
    a00 = __builtin_amdgcn_wmma_f32_16x16x32_bf16(false, fa0, false, b0, (short)0, a00, false, false);
    a01 = __builtin_amdgcn_wmma_f32_16x16x32_bf16(false, fa0, false, b1, (short)0, a01, false, false);
    a10 = __builtin_amdgcn_wmma_f32_16x16x32_bf16(false, fa1, false, b0, (short)0, a10, false, false);
    a11 = __builtin_amdgcn_wmma_f32_16x16x32_bf16(false, fa1, false, b1, (short)0, a11, false, false);
    __syncthreads();
  }
  bf16* ob = x2t + (size_t)b * D_SZ * T_SZ;
#pragma unroll
  for (int ni = 0; ni < 2; ++ni) {
    int nglob = n0 + wc.nq + ni * 16 + wc.lcol;
    float bv = day_bias[(size_t)day * D_SZ + nglob];
    bf16* col = ob + (size_t)nglob * T_SZ;
    const v8f* accs0 = ni ? &a01 : &a00;
    const v8f* accs1 = ni ? &a11 : &a10;
#pragma unroll
    for (int r = 0; r < 8; ++r) {
      int tA = t0 + wc.mq + wc.mrow + r;
      int tB = tA + 16;
      if (tA < T_SZ) col[tA] = (bf16)fast_softsign((*accs0)[r] + bv);
      if (tB < T_SZ) col[tB] = (bf16)fast_softsign((*accs1)[r] + bv);
    }
  }
}

// ---------------- kernel 3: QRNN layer-0 GEMM (gathered A, K = 2*F0) ----------
__global__ __launch_bounds__(256) void qrnn_gemm0_kernel(const bf16* __restrict__ x2t,
                                                         const bf16* __restrict__ W,
                                                         const float* __restrict__ bias,
                                                         float* __restrict__ yact) {
  __shared__ __align__(16) bf16 smA[BM * BK];
  __shared__ __align__(16) bf16 smB[BN * BK];
  const int Ktot = 2 * F0_SZ;
  const int t0 = blockIdx.x * BM, n0 = blockIdx.y * BN, b = blockIdx.z;
  const int tid = threadIdx.x;
  WaveCtx wc = wave_ctx();
  v8f a00 = {0.f,0.f,0.f,0.f,0.f,0.f,0.f,0.f};
  v8f a01 = a00, a10 = a00, a11 = a00;
  const bf16* xb = x2t + (size_t)b * D_SZ * T_SZ;
  const int ar = tid >> 1, ac = (tid & 1) * 16;
#if !HAVE_TDM
  const int brn = tid >> 2, bkg = (tid & 3) * 8;
#endif
  for (int k0 = 0; k0 < Ktot; k0 += BK) {
#if HAVE_TDM
    // B tile via Tensor Data Mover: 64 rows x 32 bf16, row stride Ktot.
    if (tid < 32)
      tdm_load_tile_2d((unsigned)(uintptr_t)smB, W + (size_t)n0 * Ktot + k0,
                       BK, BN, (unsigned)Ktot, (unsigned)G_SZ);
#else
    *(uint4*)&smB[brn * BK + bkg] =
        *(const uint4*)(W + (size_t)(n0 + brn) * Ktot + k0 + bkg);
#endif
    int tp = t0 + ar;
#pragma unroll
    for (int e = 0; e < 16; ++e) {
      int k = k0 + ac + e;
      int kk = k, tpp = tp;
      if (k >= F0_SZ) { kk = k - F0_SZ; tpp = tp - 1; }
      bf16 v = (bf16)0.0f;
      if (tpp >= 0 && tp < TP_SZ) {
        int d = kk / KLEN_;
        int kl = kk - d * KLEN_;
        v = xb[(size_t)d * T_SZ + tpp * 4 + kl];
      }
      smA[ar * BK + ac + e] = v;
    }
#if HAVE_TDM
    if (tid < 32) __builtin_amdgcn_s_wait_tensorcnt(0);
#endif
    __syncthreads();
    v16bf b0 = frag_b(smB, wc.nq);
    v16bf b1 = frag_b(smB, wc.nq + 16);
    v16bf fa0 = frag_a(smA, wc.mq);
    v16bf fa1 = frag_a(smA, wc.mq + 16);
    a00 = __builtin_amdgcn_wmma_f32_16x16x32_bf16(false, fa0, false, b0, (short)0, a00, false, false);
    a01 = __builtin_amdgcn_wmma_f32_16x16x32_bf16(false, fa0, false, b1, (short)0, a01, false, false);
    a10 = __builtin_amdgcn_wmma_f32_16x16x32_bf16(false, fa1, false, b0, (short)0, a10, false, false);
    a11 = __builtin_amdgcn_wmma_f32_16x16x32_bf16(false, fa1, false, b1, (short)0, a11, false, false);
    __syncthreads();
  }
#pragma unroll
  for (int ni = 0; ni < 2; ++ni) {
    int nglob = n0 + wc.nq + ni * 16 + wc.lcol;
    float bv = bias[nglob];
    bool is_z = nglob < H_SZ;
    const v8f* accs0 = ni ? &a01 : &a00;
    const v8f* accs1 = ni ? &a11 : &a10;
#pragma unroll
    for (int r = 0; r < 8; ++r) {
      int tpA = t0 + wc.mq + wc.mrow + r;
      int tpB = tpA + 16;
      float vA = (*accs0)[r] + bv;
      float vB = (*accs1)[r] + bv;
      vA = is_z ? fast_tanh(vA) : fast_sigmoid(vA);
      vB = is_z ? fast_tanh(vB) : fast_sigmoid(vB);
      if (tpA < TP_SZ) yact[((size_t)b * TP_SZ + tpA) * G_SZ + nglob] = vA;
      if (tpB < TP_SZ) yact[((size_t)b * TP_SZ + tpB) * G_SZ + nglob] = vB;
    }
  }
}

// ---------------- kernel 4: QRNN later-layer GEMM (K = 2*H) -------------------
__global__ __launch_bounds__(256) void qrnn_gemmR_kernel(const bf16* __restrict__ hin,
                                                         const bf16* __restrict__ W,
                                                         const float* __restrict__ bias,
                                                         float* __restrict__ yact) {
  __shared__ __align__(16) bf16 smA[BM * BK];
  __shared__ __align__(16) bf16 smB[BN * BK];
  const int Ktot = 2 * H_SZ;
  const int t0 = blockIdx.x * BM, n0 = blockIdx.y * BN, b = blockIdx.z;
  const int tid = threadIdx.x;
  WaveCtx wc = wave_ctx();
  v8f a00 = {0.f,0.f,0.f,0.f,0.f,0.f,0.f,0.f};
  v8f a01 = a00, a10 = a00, a11 = a00;
  const bf16* hb = hin + (size_t)b * TP_SZ * H_SZ;
  const int ar = tid >> 1, ac = (tid & 1) * 16;
#if !HAVE_TDM
  const int brn = tid >> 2, bkg = (tid & 3) * 8;
#endif
  for (int k0 = 0; k0 < Ktot; k0 += BK) {
#if HAVE_TDM
    if (tid < 32)
      tdm_load_tile_2d((unsigned)(uintptr_t)smB, W + (size_t)n0 * Ktot + k0,
                       BK, BN, (unsigned)Ktot, (unsigned)G_SZ);
#else
    *(uint4*)&smB[brn * BK + bkg] =
        *(const uint4*)(W + (size_t)(n0 + brn) * Ktot + k0 + bkg);
#endif
    int tp = t0 + ar;
    int k = k0 + ac;              // 16-elem group never straddles the H boundary
    int kk = k, tpp = tp;
    if (k >= H_SZ) { kk = k - H_SZ; tpp = tp - 1; }
    uint4 v0 = {0u,0u,0u,0u}, v1 = {0u,0u,0u,0u};
    if (tpp >= 0 && tp < TP_SZ) {
      const uint4* src = (const uint4*)(hb + (size_t)tpp * H_SZ + kk);
      v0 = src[0]; v1 = src[1];
    }
    uint4* dstA = (uint4*)&smA[ar * BK + ac];
    dstA[0] = v0; dstA[1] = v1;
#if HAVE_TDM
    if (tid < 32) __builtin_amdgcn_s_wait_tensorcnt(0);
#endif
    __syncthreads();
    v16bf b0 = frag_b(smB, wc.nq);
    v16bf b1 = frag_b(smB, wc.nq + 16);
    v16bf fa0 = frag_a(smA, wc.mq);
    v16bf fa1 = frag_a(smA, wc.mq + 16);
    a00 = __builtin_amdgcn_wmma_f32_16x16x32_bf16(false, fa0, false, b0, (short)0, a00, false, false);
    a01 = __builtin_amdgcn_wmma_f32_16x16x32_bf16(false, fa0, false, b1, (short)0, a01, false, false);
    a10 = __builtin_amdgcn_wmma_f32_16x16x32_bf16(false, fa1, false, b0, (short)0, a10, false, false);
    a11 = __builtin_amdgcn_wmma_f32_16x16x32_bf16(false, fa1, false, b1, (short)0, a11, false, false);
    __syncthreads();
  }
#pragma unroll
  for (int ni = 0; ni < 2; ++ni) {
    int nglob = n0 + wc.nq + ni * 16 + wc.lcol;
    float bv = bias[nglob];
    bool is_z = nglob < H_SZ;
    const v8f* accs0 = ni ? &a01 : &a00;
    const v8f* accs1 = ni ? &a11 : &a10;
#pragma unroll
    for (int r = 0; r < 8; ++r) {
      int tpA = t0 + wc.mq + wc.mrow + r;
      int tpB = tpA + 16;
      float vA = (*accs0)[r] + bv;
      float vB = (*accs1)[r] + bv;
      vA = is_z ? fast_tanh(vA) : fast_sigmoid(vA);
      vB = is_z ? fast_tanh(vB) : fast_sigmoid(vB);
      if (tpA < TP_SZ) yact[((size_t)b * TP_SZ + tpA) * G_SZ + nglob] = vA;
      if (tpB < TP_SZ) yact[((size_t)b * TP_SZ + tpB) * G_SZ + nglob] = vB;
    }
  }
}

// ---------------- kernel 5: sequential scan over Tp (c_t = f z + (1-f) c) -----
__global__ __launch_bounds__(256) void qrnn_scan_kernel(const float* __restrict__ yact,
                                                        bf16* __restrict__ hout) {
  int idx = blockIdx.x * 256 + threadIdx.x;   // b*H + h
  int b = idx >> 9;
  int h = idx & (H_SZ - 1);
  float c = 0.f;
  const float* base = yact + (size_t)b * TP_SZ * G_SZ;
  bf16* ob = hout + (size_t)b * TP_SZ * H_SZ;
  for (int tp = 0; tp < TP_SZ; ++tp) {
    const float* row = base + (size_t)tp * G_SZ;
    float z = row[h];
    float f = row[H_SZ + h];
    float o = row[2 * H_SZ + h];
    c = f * z + (1.f - f) * c;
    ob[(size_t)tp * H_SZ + h] = (bf16)(o * c);
  }
}

// ---------------- kernel 6: layernorm + 41-wide output GEMM (1 wave / pos) ----
__global__ __launch_bounds__(256) void ln_out_kernel(const bf16* __restrict__ h,
                                                     const float* __restrict__ gamma,
                                                     const float* __restrict__ beta,
                                                     const float* __restrict__ wout,
                                                     const float* __restrict__ bout,
                                                     float* __restrict__ out) {
  const int P = B_SZ * TP_SZ;
  int wave = blockIdx.x * 8 + (threadIdx.x >> 5);
  int lane = threadIdx.x & 31;
  if (wave >= P) return;
  const bf16* hp = h + (size_t)wave * H_SZ;
  float v[16];
  float s = 0.f;
#pragma unroll
  for (int i = 0; i < 16; ++i) { v[i] = (float)hp[lane + i * 32]; s += v[i]; }
  s = wave_red(s);
  float mu = s * (1.f / (float)H_SZ);
  float q = 0.f;
#pragma unroll
  for (int i = 0; i < 16; ++i) { float d = v[i] - mu; q += d * d; }
  q = wave_red(q);
  float rs = rsqrtf(q * (1.f / (float)H_SZ) + 1e-5f);
#pragma unroll
  for (int i = 0; i < 16; ++i)
    v[i] = (v[i] - mu) * rs * gamma[lane + i * 32] + beta[lane + i * 32];
  for (int c = 0; c < C_SZ; ++c) {
    const float* wr = wout + (size_t)c * H_SZ;
    float p = 0.f;
#pragma unroll
    for (int i = 0; i < 16; ++i) p += v[i] * wr[lane + i * 32];
    p = wave_red(p);
    if (lane == 0) out[(size_t)wave * C_SZ + c] = p + bout[c];
  }
}

// ---------------- orchestration ----------------------------------------------
extern "C" void kernel_launch(void* const* d_in, const int* in_sizes, int n_in,
                              void* d_out, int out_size, void* d_ws, size_t ws_size,
                              hipStream_t stream) {
  const float* neuralInput = (const float*)d_in[0];
  const int*   dayIdx      = (const int*)d_in[1];
  const float* dayWeights  = (const float*)d_in[2];
  const float* dayBias     = (const float*)d_in[3];
  const float* w0          = (const float*)d_in[4];
  const float* b0          = (const float*)d_in[5];
  const float* w_rest      = (const float*)d_in[6];
  const float* b_rest      = (const float*)d_in[7];
  const float* ln_gamma    = (const float*)d_in[8];
  const float* ln_beta     = (const float*)d_in[9];
  const float* w_out       = (const float*)d_in[10];
  const float* b_out       = (const float*)d_in[11];
  float* outp = (float*)d_out;
  (void)in_sizes; (void)n_in; (void)out_size; (void)ws_size;

  char* p = (char*)d_ws;
  auto carve = [&](size_t bytes) { char* r = p; p += (bytes + 255) & ~(size_t)255; return (void*)r; };
  const int N_XS = B_SZ * T_SZ * D_SZ;
  const int N_WD = NDAYS * D_SZ * D_SZ;
  const int N_W0 = G_SZ * 2 * F0_SZ;
  const int N_WR = 2 * G_SZ * 2 * H_SZ;
  bf16*  xs     = (bf16*)carve((size_t)N_XS * sizeof(bf16));
  bf16*  wdT_bf = (bf16*)carve((size_t)N_WD * sizeof(bf16));
  bf16*  x2t    = (bf16*)carve((size_t)N_XS * sizeof(bf16));
  bf16*  w0_bf  = (bf16*)carve((size_t)N_W0 * sizeof(bf16));
  bf16*  wr_bf  = (bf16*)carve((size_t)N_WR * sizeof(bf16));
  float* yact   = (float*)carve((size_t)B_SZ * TP_SZ * G_SZ * sizeof(float));
  bf16*  hA     = (bf16*)carve((size_t)B_SZ * TP_SZ * H_SZ * sizeof(bf16));
  bf16*  hB     = (bf16*)carve((size_t)B_SZ * TP_SZ * H_SZ * sizeof(bf16));

  cvt_bf16_kernel<<<(N_W0 + 255) / 256, 256, 0, stream>>>(w0, w0_bf, N_W0);
  cvt_bf16_kernel<<<(N_WR + 255) / 256, 256, 0, stream>>>(w_rest, wr_bf, N_WR);
  cvt_wdT_kernel<<<(N_WD + 255) / 256, 256, 0, stream>>>(dayWeights, wdT_bf);

  gauss_kernel<<<B_SZ * T_SZ, 256, 0, stream>>>(neuralInput, xs);

  {
    dim3 g((T_SZ + BM - 1) / BM, D_SZ / BN, B_SZ);
    day_gemm_kernel<<<g, 256, 0, stream>>>(xs, wdT_bf, dayBias, dayIdx, x2t);
  }

  {
    dim3 g((TP_SZ + BM - 1) / BM, G_SZ / BN, B_SZ);
    qrnn_gemm0_kernel<<<g, 256, 0, stream>>>(x2t, w0_bf, b0, yact);
  }
  qrnn_scan_kernel<<<(B_SZ * H_SZ) / 256, 256, 0, stream>>>(yact, hA);

  bf16* hin = hA;
  bf16* hout = hB;
  for (int i = 0; i < 2; ++i) {
    dim3 g((TP_SZ + BM - 1) / BM, G_SZ / BN, B_SZ);
    qrnn_gemmR_kernel<<<g, 256, 0, stream>>>(
        hin, wr_bf + (size_t)i * G_SZ * 2 * H_SZ, b_rest + (size_t)i * G_SZ, yact);
    qrnn_scan_kernel<<<(B_SZ * H_SZ) / 256, 256, 0, stream>>>(yact, hout);
    bf16* t = hin; hin = hout; hout = t;
  }

  {
    int P = B_SZ * TP_SZ;
    ln_out_kernel<<<(P + 7) / 8, 256, 0, stream>>>(hin, ln_gamma, ln_beta, w_out, b_out, outp);
  }
}
